// SoftIGNN_31044023616078
// MI455X (gfx1250) — matrix operand order repaired and checked by reading, not compiled, not run on hardware
//
#include <hip/hip_runtime.h>

// ---------------------------------------------------------------------------
// SoftIGNN forward for MI455X (gfx1250, wave32, WMMA).
//   out = relu( SpMM_gcn(edge_index, emb @ projL1(W_conv)^T) + feat @ W_mlp^T )
// GEMMs use V_WMMA_F32_16X16X32_BF16 with a hi/lo bf16 split (3 WMMAs per
// 16x16x32 step) => ~fp32 accuracy on the high-rate bf16 matrix pipe.
// SpMM is scatter-atomic: h (51MB) and out (51MB) both live in the 192MB L2,
// so edge gathers + global_atomic_add_f32 are L2-resident.
// ---------------------------------------------------------------------------

#define KAPPA_F 0.95f
#define DOUT    256
#define DIN     512

typedef __attribute__((ext_vector_type(16))) __bf16 v16bf;
typedef __attribute__((ext_vector_type(8)))  float  v8f;

__device__ __forceinline__ unsigned short f32_to_bf16(float x) {
  unsigned u = __float_as_uint(x);
  unsigned r = u + 0x7FFFu + ((u >> 16) & 1u);   // round-to-nearest-even
  return (unsigned short)(r >> 16);
}
__device__ __forceinline__ float bf16_to_f32(unsigned short h) {
  return __uint_as_float(((unsigned)h) << 16);
}

// ---------------------------------------------------------------------------
// Kernel 1: row-wise l1-ball projection of W_conv (256 x 256), 1 block/row.
// ---------------------------------------------------------------------------
__global__ __launch_bounds__(256)
void ignn_proj_kernel(const float* __restrict__ W, float* __restrict__ Wc) {
  __shared__ float s[256];      // sorted |w|
  __shared__ float red[256];    // reductions / scan
  __shared__ float cssArr[256];
  __shared__ float l1Sh, thetaSh;

  const int r = blockIdx.x;
  const int t = threadIdx.x;
  const float w = W[r * 256 + t];
  const float a = fabsf(w);

  // row l1 norm
  red[t] = a; __syncthreads();
  #pragma unroll
  for (int off = 128; off > 0; off >>= 1) {
    if (t < off) red[t] += red[t + off];
    __syncthreads();
  }
  if (t == 0) l1Sh = red[0];
  s[t] = a;
  __syncthreads();

  // bitonic sort, descending
  for (int k = 2; k <= 256; k <<= 1) {
    for (int j = k >> 1; j > 0; j >>= 1) {
      int ixj = t ^ j;
      if (ixj > t) {
        float x = s[t], y = s[ixj];
        bool desc = ((t & k) == 0);
        if (desc ? (x < y) : (x > y)) { s[t] = y; s[ixj] = x; }
      }
      __syncthreads();
    }
  }

  // inclusive prefix sum of sorted values
  red[t] = s[t]; __syncthreads();
  for (int off = 1; off < 256; off <<= 1) {
    float v = (t >= off) ? red[t - off] : 0.f;
    __syncthreads();
    red[t] += v;
    __syncthreads();
  }
  float css = red[t] - KAPPA_F;
  cssArr[t] = css;
  float p = (s[t] * (float)(t + 1) > css) ? 1.f : 0.f;
  __syncthreads();
  red[t] = p; __syncthreads();
  #pragma unroll
  for (int off = 128; off > 0; off >>= 1) {
    if (t < off) red[t] += red[t + off];
    __syncthreads();
  }
  if (t == 0) {
    int rho = (int)red[0];
    thetaSh = cssArr[rho - 1] / (float)rho;
  }
  __syncthreads();

  float outv = (l1Sh > KAPPA_F) ? copysignf(fmaxf(a - thetaSh, 0.f), w) : w;
  Wc[r * 256 + t] = outv;
}

// ---------------------------------------------------------------------------
// Kernel 2: tiled GEMM  C[nrows x 256] = A[nrows x K] * W[256 x K]^T
// bf16 hi/lo split, 3x V_WMMA_F32_16X16X32_BF16 per 16x16x32 step.
// Block: 256 threads = 8 waves (4 m-waves x 2 n-waves), wave tile 32x32,
// block tile 128x64, K-chunk 32 staged through LDS (padded, stride 34).
// ---------------------------------------------------------------------------
#define BM 128
#define BN 64
#define BK 32
#define LDA 34
#define LDB 34

__global__ __launch_bounds__(256)
void ignn_gemm_bf16x3_kernel(const float* __restrict__ A,
                             const float* __restrict__ W,
                             float* __restrict__ C, int nrows, int K) {
  __shared__ unsigned short Ah[BM * LDA];
  __shared__ unsigned short Al[BM * LDA];
  __shared__ unsigned short Bh[BN * LDB];
  __shared__ unsigned short Bl[BN * LDB];

  const int tid   = threadIdx.x;
  const int lane  = tid & 31;
  const int wid   = tid >> 5;
  const int mBase = (wid >> 1) * 32;   // 4 m-waves
  const int nBase = (wid & 1) * 32;    // 2 n-waves
  const int rowBlock = blockIdx.x * BM;
  const int colBlock = blockIdx.y * BN;

  v8f acc[2][2];
  #pragma unroll
  for (int a = 0; a < 2; ++a)
    #pragma unroll
    for (int b = 0; b < 2; ++b)
      #pragma unroll
      for (int j = 0; j < 8; ++j) acc[a][b][j] = 0.f;

  // ISA 7.12.2 layouts (wave32):
  //  A 16x32 bf16: lane<16 -> M=lane, VGPR j holds K pair at {0,2,4,6,16,18,20,22}[j];
  //                lane>=16 -> same +8.
  //  B 32x16 bf16: lane<16 -> N=lane, VGPR j holds K pair at {0,2,4,6,16,18,20,22}[j]; lane>=16 -> same +8.
  const int base8 = (lane >> 4) * 8;
  const int kbB   = (lane >> 4) * 16;
  const int mlane = lane & 15;

  for (int k0 = 0; k0 < K; k0 += BK) {
    __syncthreads();
    // ---- stage A: 128x32 f32 = 1024 float4, 4 per thread, split hi/lo bf16
    #pragma unroll
    for (int it = 0; it < 4; ++it) {
      int fidx = tid + it * 256;
      int row  = fidx >> 3;
      int c4   = (fidx & 7) * 4;
      int gr   = rowBlock + row;
      float4 v = make_float4(0.f, 0.f, 0.f, 0.f);
      if (gr < nrows) {
        v = *(const float4*)(A + (size_t)gr * K + k0 + c4);
        if (k0 + BK < K)
          __builtin_prefetch(A + (size_t)gr * K + k0 + BK + c4, 0, 1);
      }
      unsigned short h0 = f32_to_bf16(v.x), h1 = f32_to_bf16(v.y);
      unsigned short h2 = f32_to_bf16(v.z), h3 = f32_to_bf16(v.w);
      unsigned short l0 = f32_to_bf16(v.x - bf16_to_f32(h0));
      unsigned short l1 = f32_to_bf16(v.y - bf16_to_f32(h1));
      unsigned short l2 = f32_to_bf16(v.z - bf16_to_f32(h2));
      unsigned short l3 = f32_to_bf16(v.w - bf16_to_f32(h3));
      *(unsigned*)&Ah[row * LDA + c4]     = (unsigned)h0 | ((unsigned)h1 << 16);
      *(unsigned*)&Ah[row * LDA + c4 + 2] = (unsigned)h2 | ((unsigned)h3 << 16);
      *(unsigned*)&Al[row * LDA + c4]     = (unsigned)l0 | ((unsigned)l1 << 16);
      *(unsigned*)&Al[row * LDA + c4 + 2] = (unsigned)l2 | ((unsigned)l3 << 16);
    }
    // ---- stage B: 64x32 f32 = 512 float4, 2 per thread (W is [256 x K] row-major)
    #pragma unroll
    for (int it = 0; it < 2; ++it) {
      int fidx = tid + it * 256;
      int row  = fidx >> 3;                 // 0..63 = output column
      int c4   = (fidx & 7) * 4;
      float4 v = *(const float4*)(W + (size_t)(colBlock + row) * K + k0 + c4);
      unsigned short h0 = f32_to_bf16(v.x), h1 = f32_to_bf16(v.y);
      unsigned short h2 = f32_to_bf16(v.z), h3 = f32_to_bf16(v.w);
      unsigned short l0 = f32_to_bf16(v.x - bf16_to_f32(h0));
      unsigned short l1 = f32_to_bf16(v.y - bf16_to_f32(h1));
      unsigned short l2 = f32_to_bf16(v.z - bf16_to_f32(h2));
      unsigned short l3 = f32_to_bf16(v.w - bf16_to_f32(h3));
      *(unsigned*)&Bh[row * LDB + c4]     = (unsigned)h0 | ((unsigned)h1 << 16);
      *(unsigned*)&Bh[row * LDB + c4 + 2] = (unsigned)h2 | ((unsigned)h3 << 16);
      *(unsigned*)&Bl[row * LDB + c4]     = (unsigned)l0 | ((unsigned)l1 << 16);
      *(unsigned*)&Bl[row * LDB + c4 + 2] = (unsigned)l2 | ((unsigned)l3 << 16);
    }
    __syncthreads();

    union Frag { v16bf v; unsigned u[8]; };
    Frag fah[2], fal[2], fbh[2], fbl[2];
    const int koffA[8] = {0, 2, 4, 6, 16, 18, 20, 22};
    #pragma unroll
    for (int mi = 0; mi < 2; ++mi) {
      int m = mBase + mi * 16 + mlane;
      #pragma unroll
      for (int j = 0; j < 8; ++j) {
        int ko = base8 + koffA[j];
        fah[mi].u[j] = *(const unsigned*)&Ah[m * LDA + ko];
        fal[mi].u[j] = *(const unsigned*)&Al[m * LDA + ko];
      }
    }
    #pragma unroll
    for (int ni = 0; ni < 2; ++ni) {
      int n = nBase + ni * 16 + mlane;
      #pragma unroll
      for (int j = 0; j < 8; ++j) {
        int ko = base8 + koffA[j];
        fbh[ni].u[j] = *(const unsigned*)&Bh[n * LDB + ko];
        fbl[ni].u[j] = *(const unsigned*)&Bl[n * LDB + ko];
      }
    }
    #pragma unroll
    for (int mi = 0; mi < 2; ++mi)
      #pragma unroll
      for (int ni = 0; ni < 2; ++ni) {
        acc[mi][ni] = __builtin_amdgcn_wmma_f32_16x16x32_bf16(
            false, fah[mi].v, false, fbh[ni].v, (short)0, acc[mi][ni], false, false);
        acc[mi][ni] = __builtin_amdgcn_wmma_f32_16x16x32_bf16(
            false, fah[mi].v, false, fbl[ni].v, (short)0, acc[mi][ni], false, false);
        acc[mi][ni] = __builtin_amdgcn_wmma_f32_16x16x32_bf16(
            false, fal[mi].v, false, fbh[ni].v, (short)0, acc[mi][ni], false, false);
      }
  }

  // epilogue: C/D layout — VGPR j: lanes 0-15 -> (M=j, N=lane), lanes 16-31 -> (M=j+8)
  const int rowAdd = (lane >> 4) * 8;
  #pragma unroll
  for (int mi = 0; mi < 2; ++mi)
    #pragma unroll
    for (int ni = 0; ni < 2; ++ni)
      #pragma unroll
      for (int j = 0; j < 8; ++j) {
        int gr = rowBlock + mBase + mi * 16 + rowAdd + j;
        int gc = colBlock + nBase + ni * 16 + mlane;
        if (gr < nrows) C[(size_t)gr * DOUT + gc] = acc[mi][ni][j];
      }
}

// ---------------------------------------------------------------------------
// GCN helpers
// ---------------------------------------------------------------------------
__global__ void ignn_deg_init_kernel(int* deg, int n) {
  int i = blockIdx.x * blockDim.x + threadIdx.x;
  if (i < n) deg[i] = 1;                       // self-loop contribution
}
__global__ void ignn_deg_count_kernel(const int* __restrict__ col, int* deg, int e) {
  int i = blockIdx.x * blockDim.x + threadIdx.x;
  if (i < e) atomicAdd(&deg[col[i]], 1);
}
__global__ void ignn_dinv_kernel(const int* __restrict__ deg, float* dinv, int n) {
  int i = blockIdx.x * blockDim.x + threadIdx.x;
  if (i < n) dinv[i] = rsqrtf((float)deg[i]);
}

// ---------------------------------------------------------------------------
// SpMM scatter: one edge per wave; 32 lanes x 8 floats cover the 256-wide row.
// h and out are L2-resident (51MB each << 192MB L2).
// ---------------------------------------------------------------------------
__global__ __launch_bounds__(256)
void ignn_spmm_kernel(const int* __restrict__ ei, const float* __restrict__ dinv,
                      const float* __restrict__ h, float* __restrict__ out,
                      int E, int ET) {
  const int wid  = threadIdx.x >> 5;
  const int lane = threadIdx.x & 31;
  const int e = blockIdx.x * 8 + wid;
  if (e >= ET) return;
  int r, c;
  if (e < E) { r = ei[e]; c = ei[E + e]; }
  else       { r = e - E; c = r; }
  const float norm = dinv[r] * dinv[c];
  const float4* hs = (const float4*)(h + (size_t)r * DOUT) + lane * 2;
  float4 x = hs[0], y = hs[1];
  float* o = out + (size_t)c * DOUT + lane * 8;
  unsafeAtomicAdd(o + 0, norm * x.x);
  unsafeAtomicAdd(o + 1, norm * x.y);
  unsafeAtomicAdd(o + 2, norm * x.z);
  unsafeAtomicAdd(o + 3, norm * x.w);
  unsafeAtomicAdd(o + 4, norm * y.x);
  unsafeAtomicAdd(o + 5, norm * y.y);
  unsafeAtomicAdd(o + 6, norm * y.z);
  unsafeAtomicAdd(o + 7, norm * y.w);
}

__global__ void ignn_relu_kernel(float* out, int total) {
  int i = blockIdx.x * blockDim.x + threadIdx.x;
  if (i < total) out[i] = fmaxf(out[i], 0.f);
}

// ---------------------------------------------------------------------------
extern "C" void kernel_launch(void* const* d_in, const int* in_sizes, int n_in,
                              void* d_out, int out_size, void* d_ws, size_t ws_size,
                              hipStream_t stream) {
  const float* features   = (const float*)d_in[0];
  const int*   edge_index = (const int*)  d_in[1];
  const float* embeddings = (const float*)d_in[2];
  const float* W_conv     = (const float*)d_in[3];
  const float* W_mlp      = (const float*)d_in[4];
  float* out = (float*)d_out;

  const int N  = in_sizes[0] / DIN;     // 50000
  const int E  = in_sizes[1] / 2;       // 1600000
  const int ET = E + N;

  // workspace carve: h (N*256 f32), Wc (256*256 f32), deg (N i32), dinv (N f32)
  char* ws = (char*)d_ws;
  size_t off = 0;
  float* h    = (float*)(ws + off); off += (size_t)N * DOUT * sizeof(float);
  float* Wc   = (float*)(ws + off); off += (size_t)DOUT * DOUT * sizeof(float);
  int*   deg  = (int*)  (ws + off); off += ((size_t)N * sizeof(int) + 255) & ~(size_t)255;
  float* dinv = (float*)(ws + off); off += ((size_t)N * sizeof(float) + 255) & ~(size_t)255;

  // 1. project W_conv rows onto l1 ball
  ignn_proj_kernel<<<DOUT, 256, 0, stream>>>(W_conv, Wc);
  // 2. degrees (init=1 self-loop) and dinv
  ignn_deg_init_kernel<<<(N + 255) / 256, 256, 0, stream>>>(deg, N);
  ignn_deg_count_kernel<<<(E + 255) / 256, 256, 0, stream>>>(edge_index + E, deg, E);
  ignn_dinv_kernel<<<(N + 255) / 256, 256, 0, stream>>>(deg, dinv, N);
  // 3. h = embeddings @ Wc^T   (K = 256)
  dim3 g1((N + BM - 1) / BM, DOUT / BN);
  ignn_gemm_bf16x3_kernel<<<g1, 256, 0, stream>>>(embeddings, Wc, h, N, DOUT);
  // 4. out = features @ W_mlp^T (K = 512) — full overwrite of d_out
  ignn_gemm_bf16x3_kernel<<<g1, 256, 0, stream>>>(features, W_mlp, out, N, DIN);
  // 5. out[col] += norm * h[row]
  ignn_spmm_kernel<<<(ET + 7) / 8, 256, 0, stream>>>(edge_index, dinv, h, out, E, ET);
  // 6. relu
  ignn_relu_kernel<<<((size_t)N * DOUT + 255) / 256, 256, 0, stream>>>(out, N * DOUT);
}
